// TextConditionedViTKV_18433999635171
// MI455X (gfx1250) — compile-verified, matrix operand order checked by reading, not run on hardware
//
#include <hip/hip_runtime.h>
#include <hip/hip_bf16.h>

// ---------------- types ----------------
typedef __bf16 bf16;
typedef __attribute__((ext_vector_type(16))) __bf16 bf16x16;
typedef __attribute__((ext_vector_type(8)))  __bf16 bf16x8;
typedef __attribute__((ext_vector_type(8)))  float  f32x8;

__device__ __forceinline__ f32x8 wmma_bf16(bf16x16 a, bf16x16 b, f32x8 c) {
    // D = A(16x32 bf16) * B(32x16 bf16) + C(16x16 f32)
    return __builtin_amdgcn_wmma_f32_16x16x32_bf16(
        false, a, false, b, (short)0, c, false, false);
}

__device__ __forceinline__ f32x8 zero8() {
    f32x8 z;
#pragma unroll
    for (int i = 0; i < 8; ++i) z[i] = 0.0f;
    return z;
}

// Load an A/B fragment from a K-major bf16 row.
// p = row start + k-block base (row chosen per lane as lane&15).
// Lane<16  : elems 0..7 -> K {0..7},  elems 8..15 -> K {16..23}
// Lane>=16 : elems 0..7 -> K {8..15}, elems 8..15 -> K {24..31}
__device__ __forceinline__ bf16x16 load_frag(const bf16* p) {
    const int half = (threadIdx.x & 16) ? 8 : 0;
    bf16x8 lo = *(const bf16x8*)(p + half);
    bf16x8 hi = *(const bf16x8*)(p + half + 16);
    bf16x16 f;
#pragma unroll
    for (int j = 0; j < 8; ++j) { f[j] = lo[j]; f[j + 8] = hi[j]; }
    return f;
}

// Async copy of one 128B V row per lane into LDS (8 x b128, ASYNCcnt-tracked).
// INST_OFFSET is added on both the LDS and global side, so one base each suffices.
__device__ __forceinline__ void issue_v_async(const bf16* src, const bf16* dstlds) {
    unsigned l = (unsigned)(size_t)dstlds;          // low 32 bits of LDS aperture addr
    unsigned long long g = (unsigned long long)(size_t)src;
    asm volatile(
        "global_load_async_to_lds_b128 %0, %1, off\n\t"
        "global_load_async_to_lds_b128 %0, %1, off offset:16\n\t"
        "global_load_async_to_lds_b128 %0, %1, off offset:32\n\t"
        "global_load_async_to_lds_b128 %0, %1, off offset:48\n\t"
        "global_load_async_to_lds_b128 %0, %1, off offset:64\n\t"
        "global_load_async_to_lds_b128 %0, %1, off offset:80\n\t"
        "global_load_async_to_lds_b128 %0, %1, off offset:96\n\t"
        "global_load_async_to_lds_b128 %0, %1, off offset:112"
        :: "v"(l), "v"(g) : "memory");
}

// ---------------- fp32 -> bf16 convert ----------------
__global__ void cvt_bf16_kernel(const float* __restrict__ x, bf16* __restrict__ y, size_t n) {
    size_t i = (size_t)blockIdx.x * blockDim.x + threadIdx.x;
    if (i < n) y[i] = (bf16)x[i];
}

// ---------------- generic WMMA GEMM: Y = X * W^T + b ----------------
// X: MxK bf16 row-major, W: NxK bf16 row-major. One wave computes a 32x64 tile:
// 2 A-frags x 4 B-frags -> 8 WMMAs per 32-wide K step (each B reused twice).
// K and MODE are compile-time: all 6 fragment rows address off just 2 base
// pointers via immediate offsets (n*16*K*2 <= 240KB < 8MB imm range) -> no spills.
// MODE 0: bf16 out; 1: GELU(exact) then bf16; 2: f32 out; 3: *tanh(gate[col/64]) then bf16
template <int K, int MODE>
__global__ __launch_bounds__(256, 1)
void gemm_bf16_kernel(const bf16* __restrict__ X, const bf16* __restrict__ W,
                      const float* __restrict__ bias, void* __restrict__ Yv,
                      int M, int N, const float* __restrict__ gate) {
    const int wavesPerBlk = blockDim.x >> 5;
    int wave = blockIdx.x * wavesPerBlk + (threadIdx.x >> 5);
    const int Mt = M >> 5, Nt = N >> 6;
    if (wave >= Mt * Nt) return;                 // wave-uniform: EXEC stays full
    const int tm = wave / Nt, tn = wave % Nt;
    const int lane = threadIdx.x & 31;
    const int r16  = lane & 15;

    const bf16* xrow = X + (size_t)(tm * 32 + r16) * K;   // row1 = xrow + 16*K (imm)
    const bf16* wrow = W + (size_t)(tn * 64 + r16) * K;   // rows n at wrow + n*16*K (imm)

    f32x8 acc[2][4];
#pragma unroll
    for (int mt = 0; mt < 2; ++mt)
#pragma unroll
        for (int n = 0; n < 4; ++n) acc[mt][n] = zero8();

    for (int k = 0; k < K; k += 32) {
        // prefetch two K-steps ahead (global_prefetch_b8; speculative, fault-safe)
        __builtin_prefetch(xrow + k + 64, 0, 3);
        __builtin_prefetch(xrow + 16 * K + k + 64, 0, 3);
#pragma unroll
        for (int n = 0; n < 4; ++n)
            __builtin_prefetch(wrow + n * 16 * K + k + 64, 0, 3);

        bf16x16 a0 = load_frag(xrow + k);
        bf16x16 a1 = load_frag(xrow + 16 * K + k);
#pragma unroll
        for (int n = 0; n < 4; ++n) {
            bf16x16 b = load_frag(wrow + n * 16 * K + k);
            acc[0][n] = wmma_bf16(a0, b, acc[0][n]);
            acc[1][n] = wmma_bf16(a1, b, acc[1][n]);
        }
    }

#pragma unroll
    for (int mt = 0; mt < 2; ++mt) {
        const int rowbase = tm * 32 + mt * 16 + ((lane & 16) ? 8 : 0);
#pragma unroll
        for (int n = 0; n < 4; ++n) {
            const int col = tn * 64 + n * 16 + r16;
            const float bcol = bias[col];
            float g = 1.0f;
            if constexpr (MODE == 3) g = tanhf(gate[col >> 6]);
#pragma unroll
            for (int r = 0; r < 8; ++r) {
                float y = acc[mt][n][r] + bcol;
                if constexpr (MODE == 1) y = 0.5f * y * (1.0f + erff(y * 0.70710678118654752f));
                y *= g;
                const size_t idx = (size_t)(rowbase + r) * N + col;
                if constexpr (MODE == 2) ((float*)Yv)[idx] = y;
                else                     ((bf16*)Yv)[idx]  = (bf16)y;
            }
        }
    }
}

// ---------------- LayerNorm over rows of 256 (ctx) ----------------
__global__ void ln256_kernel(const float* __restrict__ x, const float* __restrict__ g,
                             const float* __restrict__ b, bf16* __restrict__ y, int T) {
    int row = blockIdx.x * (blockDim.x >> 5) + (threadIdx.x >> 5);
    if (row >= T) return;
    const int lane = threadIdx.x & 31;
    const float* p = x + (size_t)row * 256;
    float s = 0.f, s2 = 0.f;
    for (int j = lane; j < 256; j += 32) { float v = p[j]; s += v; s2 += v * v; }
#pragma unroll
    for (int off = 16; off >= 1; off >>= 1) {
        s  += __shfl_xor(s,  off, 32);
        s2 += __shfl_xor(s2, off, 32);
    }
    const float mu  = s * (1.0f / 256.0f);
    const float var = s2 * (1.0f / 256.0f) - mu * mu;
    const float inv = rsqrtf(var + 1e-5f);
    for (int j = lane; j < 256; j += 32)
        y[(size_t)row * 256 + j] = (bf16)((p[j] - mu) * inv * g[j] + b[j]);
}

// ---------------- RoPE on q and k halves of packed qkv (bf16, in-place) ----------------
__global__ void rope_kernel(bf16* __restrict__ qkv, const float* __restrict__ cosb,
                            const float* __restrict__ sinb, int S) {
    size_t i = (size_t)blockIdx.x * blockDim.x + threadIdx.x;
    const size_t total = (size_t)S * 2 * 16 * 32;   // [S][part][H][D/2]
    if (i >= total) return;
    const int d    = (int)(i & 31);
    const int h    = (int)((i >> 5) & 15);
    const int part = (int)((i >> 9) & 1);
    const int s    = (int)(i >> 10);
    bf16* base = qkv + (size_t)s * 3072 + part * 1024 + h * 64;
    const float x1 = (float)base[d], x2 = (float)base[d + 32];
    const float c1 = cosb[s * 64 + d],      s1 = sinb[s * 64 + d];
    const float c2 = cosb[s * 64 + d + 32], s2 = sinb[s * 64 + d + 32];
    base[d]      = (bf16)(x1 * c1 - x2 * s1);   // x*cos + (-x2)*sin
    base[d + 32] = (bf16)(x2 * c2 + x1 * s2);   // x*cos + ( x1)*sin
}

// per-lane V row pointer for kv block starting at j0 (block-aligned vs 1024 boundary)
__device__ __forceinline__ const bf16* vrow_ptr(const bf16* qkv, const bf16* vtext,
                                                int c, int h, int j0, int lane) {
    if (j0 < 1024) return qkv   + (size_t)(c * 1024 + j0 + lane)        * 3072 + 2048 + h * 64;
    return               vtext  + (size_t)(c * 64 + (j0 - 1024) + lane) * 1024 + h * 64;
}

// ---------------- flash attention: one wave per (chunk, head, 16-row q tile) ----------------
// qkv: [S][3*1024] bf16 (q|k|v each [H][64]); ktext/vtext: [T][1024] bf16.
// kv length per chunk = 1024 + 64 = 1088 = 34 blocks of 32 (boundary block-aligned).
// V blocks are double-buffered in LDS via GLOBAL_LOAD_ASYNC_TO_LDS_B128 (ASYNCcnt).
// launch_bounds(32,1): relax the register cap so Q fragments stay resident (no
// scratch reloads in the kv loop); latency is hidden by the async V pipeline.
__global__ __launch_bounds__(32, 1)
void attn_kernel(const bf16* __restrict__ qkv, const bf16* __restrict__ ktext,
                 const bf16* __restrict__ vtext, bf16* __restrict__ out) {
    __shared__ __align__(16) bf16 plds[16 * 32];
    __shared__ __align__(16) bf16 vlds[2][32 * 64];

    const int lane = threadIdx.x;
    const int r16  = lane & 15;
    int tile = blockIdx.x;
    const int qt = tile & 63;           // 64 q tiles per (c,h)
    const int h  = (tile >> 6) & 15;
    const int c  = tile >> 10;

    const bf16* qbase = qkv + (size_t)(c * 1024 + qt * 16 + r16) * 3072 + h * 64;
    const bf16x16 qa0 = load_frag(qbase);
    const bf16x16 qa1 = load_frag(qbase + 32);

    float m[8], l[8];
    f32x8 o[4];
#pragma unroll
    for (int r = 0; r < 8; ++r) { m[r] = -3.0e38f; l[r] = 0.0f; }
#pragma unroll
    for (int n = 0; n < 4; ++n) o[n] = zero8();

    const float scale = 0.125f;         // 1/sqrt(64)

    // prologue: kick off async V load for block 0
    issue_v_async(vrow_ptr(qkv, vtext, c, h, 0, lane), &vlds[0][lane * 64]);

    for (int kb = 0; kb < 34; ++kb) {
        const int j0 = kb * 32;
        // ---- scores for 32 kv cols: two 16x16 C fragments (overlaps pending async V)
        const bf16 *kr0, *kr1;
        if (kb < 32) {
            kr0 = qkv + (size_t)(c * 1024 + j0 + r16)      * 3072 + 1024 + h * 64;
            kr1 = qkv + (size_t)(c * 1024 + j0 + 16 + r16) * 3072 + 1024 + h * 64;
        } else {
            const int jt = j0 - 1024;
            kr0 = ktext + (size_t)(c * 64 + jt + r16)      * 1024 + h * 64;
            kr1 = ktext + (size_t)(c * 64 + jt + 16 + r16) * 1024 + h * 64;
        }
        f32x8 s0 = zero8(), s1 = zero8();
        s0 = wmma_bf16(qa0, load_frag(kr0),      s0);
        s0 = wmma_bf16(qa1, load_frag(kr0 + 32), s0);
        s1 = wmma_bf16(qa0, load_frag(kr1),      s1);
        s1 = wmma_bf16(qa1, load_frag(kr1 + 32), s1);

        // ---- streaming softmax update (rows live in 16-lane halves)
        float mx[8];
#pragma unroll
        for (int r = 0; r < 8; ++r) {
            s0[r] *= scale; s1[r] *= scale;
            mx[r] = fmaxf(s0[r], s1[r]);
        }
#pragma unroll
        for (int off = 8; off >= 1; off >>= 1)
#pragma unroll
            for (int r = 0; r < 8; ++r) mx[r] = fmaxf(mx[r], __shfl_xor(mx[r], off, 32));

        float p0[8], p1[8], alpha[8], ps[8];
#pragma unroll
        for (int r = 0; r < 8; ++r) {
            const float mn = fmaxf(m[r], mx[r]);
            alpha[r] = __expf(m[r] - mn);
            m[r] = mn;
            p0[r] = __expf(s0[r] - mn);
            p1[r] = __expf(s1[r] - mn);
            ps[r] = p0[r] + p1[r];
        }
#pragma unroll
        for (int off = 8; off >= 1; off >>= 1)
#pragma unroll
            for (int r = 0; r < 8; ++r) ps[r] += __shfl_xor(ps[r], off, 32);
#pragma unroll
        for (int r = 0; r < 8; ++r) l[r] = l[r] * alpha[r] + ps[r];
#pragma unroll
        for (int n = 0; n < 4; ++n)
#pragma unroll
            for (int r = 0; r < 8; ++r) o[n][r] *= alpha[r];

        // ---- C-layout P -> LDS -> A-layout fragment
        {
            const int rb = (lane & 16) ? 8 : 0;
#pragma unroll
            for (int r = 0; r < 8; ++r) {
                plds[(rb + r) * 32 + r16]      = (bf16)p0[r];
                plds[(rb + r) * 32 + r16 + 16] = (bf16)p1[r];
            }
        }
        __syncthreads();
        const bf16x16 pa = load_frag(plds + r16 * 32);

        // ---- wait for this block's async V, then O += P * V
        asm volatile("s_wait_asynccnt 0x0" ::: "memory");
        const bf16* vb_base = vlds[kb & 1];
#pragma unroll
        for (int n = 0; n < 4; ++n) {
            bf16x16 vb;
            const int d  = n * 16 + r16;
            const int kh = (lane & 16) ? 8 : 0;
#pragma unroll
            for (int j = 0; j < 16; ++j) {
                const int kk = ((j < 8) ? j : j + 8) + kh;
                vb[j] = vb_base[kk * 64 + d];
            }
            o[n] = wmma_bf16(pa, vb, o[n]);
        }

        // ---- kick off next block's async V into the other buffer
        if (kb < 33) {
            asm volatile("s_wait_dscnt 0x0" ::: "memory");   // ds gathers done before overwrite
            issue_v_async(vrow_ptr(qkv, vtext, c, h, j0 + 32, lane),
                          &vlds[(kb + 1) & 1][lane * 64]);
        }
        __syncthreads();
    }

    // ---- normalize and store (bf16, [S][H*64] layout)
    const int rb = (lane & 16) ? 8 : 0;
#pragma unroll
    for (int n = 0; n < 4; ++n)
#pragma unroll
        for (int r = 0; r < 8; ++r) {
            const size_t row = (size_t)(c * 1024 + qt * 16 + rb + r);
            const int col = h * 64 + n * 16 + r16;
            out[row * 1024 + col] = (bf16)(o[n][r] / l[r]);
        }
}

// ---------------- host-side templated GEMM launcher ----------------
template <int K, int MODE>
static void launch_gemm(const bf16* X, const bf16* W, const float* bias, void* Y,
                        int M, int N, const float* gate, hipStream_t stream) {
    int waves = (M >> 5) * (N >> 6);          // 32x64 tile per wave
    int blocks = (waves + 7) / 8;
    gemm_bf16_kernel<K, MODE><<<blocks, 256, 0, stream>>>(X, W, bias, Y, M, N, gate);
}

// ---------------- host launcher ----------------
extern "C" void kernel_launch(void* const* d_in, const int* in_sizes, int n_in,
                              void* d_out, int out_size, void* d_ws, size_t ws_size,
                              hipStream_t stream) {
    const float* hidden = (const float*)d_in[0];
    const float* text   = (const float*)d_in[1];
    const float* cosb   = (const float*)d_in[2];
    const float* sinb   = (const float*)d_in[3];
    const float* w_qkv  = (const float*)d_in[4];
    const float* b_qkv  = (const float*)d_in[5];
    const float* w_proj = (const float*)d_in[6];
    const float* b_proj = (const float*)d_in[7];
    const float* w1     = (const float*)d_in[8];
    const float* b1     = (const float*)d_in[9];
    const float* w2     = (const float*)d_in[10];
    const float* b2     = (const float*)d_in[11];
    const float* ln_g   = (const float*)d_in[12];
    const float* ln_b   = (const float*)d_in[13];
    const float* w_k    = (const float*)d_in[14];
    const float* b_k    = (const float*)d_in[15];
    const float* w_v    = (const float*)d_in[16];
    const float* b_v    = (const float*)d_in[17];
    const float* gate   = (const float*)d_in[18];
    float* outp = (float*)d_out;

    const int S = 8192, DIM = 1024, T = 512, TDIM = 2560, CTX = 256;

    // workspace layout (256B aligned)
    size_t off = 0;
    auto carve = [&](size_t bytes) { size_t o = off; off += (bytes + 255) & ~(size_t)255; return o; };
    char* ws = (char*)d_ws;
    bf16* hid_bf   = (bf16*)(ws + carve((size_t)S * DIM * 2));
    bf16* wqkv_bf  = (bf16*)(ws + carve((size_t)3 * DIM * DIM * 2));
    bf16* wproj_bf = (bf16*)(ws + carve((size_t)DIM * DIM * 2));
    bf16* w1_bf    = (bf16*)(ws + carve((size_t)2 * CTX * TDIM * 2));
    bf16* w2_bf    = (bf16*)(ws + carve((size_t)CTX * 2 * CTX * 2));
    bf16* wk_bf    = (bf16*)(ws + carve((size_t)DIM * CTX * 2));
    bf16* wv_bf    = (bf16*)(ws + carve((size_t)DIM * CTX * 2));
    bf16* txt_bf   = (bf16*)(ws + carve((size_t)T * TDIM * 2));
    bf16* h_bf     = (bf16*)(ws + carve((size_t)T * 2 * CTX * 2));
    float* ctx_f32 = (float*)(ws + carve((size_t)T * CTX * 4));
    bf16* ctx_bf   = (bf16*)(ws + carve((size_t)T * CTX * 2));
    bf16* kt_bf    = (bf16*)(ws + carve((size_t)T * DIM * 2));
    bf16* vt_bf    = (bf16*)(ws + carve((size_t)T * DIM * 2));
    bf16* qkv_bf   = (bf16*)(ws + carve((size_t)S * 3 * DIM * 2));
    bf16* att_bf   = (bf16*)(ws + carve((size_t)S * DIM * 2));
    (void)ws_size; (void)n_in; (void)in_sizes; (void)out_size;

    auto cvt = [&](const float* src, bf16* dst, size_t n) {
        int blocks = (int)((n + 255) / 256);
        cvt_bf16_kernel<<<blocks, 256, 0, stream>>>(src, dst, n);
    };

    // 1) bf16 conversions
    cvt(hidden, hid_bf,  (size_t)S * DIM);
    cvt(w_qkv,  wqkv_bf, (size_t)3 * DIM * DIM);
    cvt(w_proj, wproj_bf,(size_t)DIM * DIM);
    cvt(w1,     w1_bf,   (size_t)2 * CTX * TDIM);
    cvt(w2,     w2_bf,   (size_t)CTX * 2 * CTX);
    cvt(w_k,    wk_bf,   (size_t)DIM * CTX);
    cvt(w_v,    wv_bf,   (size_t)DIM * CTX);
    cvt(text,   txt_bf,  (size_t)T * TDIM);

    // 2) text MLP: h = gelu(text @ w1^T + b1)       (512x512 K=2560)
    launch_gemm<2560, 1>(txt_bf, w1_bf, b1, h_bf, T, 2 * CTX, nullptr, stream);
    // 3) ctx = h @ w2^T + b2 (f32)                  (512x256 K=512)
    launch_gemm<512, 2>(h_bf, w2_bf, b2, ctx_f32, T, CTX, nullptr, stream);
    // 4) LayerNorm -> bf16
    ln256_kernel<<<T / 8, 256, 0, stream>>>(ctx_f32, ln_g, ln_b, ctx_bf, T);
    // 5) k_text / v_text with tanh-gate epilogue    (512x1024 K=256)
    launch_gemm<256, 3>(ctx_bf, wk_bf, b_k, kt_bf, T, DIM, gate, stream);
    launch_gemm<256, 3>(ctx_bf, wv_bf, b_v, vt_bf, T, DIM, gate, stream);
    // 6) qkv = hidden @ w_qkv^T + b_qkv             (8192x3072 K=1024)
    launch_gemm<1024, 0>(hid_bf, wqkv_bf, b_qkv, qkv_bf, S, 3 * DIM, nullptr, stream);
    // 7) RoPE on q,k (in place, bf16)
    {
        size_t total = (size_t)S * 2 * 16 * 32;
        rope_kernel<<<(int)((total + 255) / 256), 256, 0, stream>>>(qkv_bf, cosb, sinb, S);
    }
    // 8) flash attention: 8 chunks x 16 heads x 64 q-tiles = 8192 waves
    attn_kernel<<<8192, 32, 0, stream>>>(qkv_bf, kt_bf, vt_bf, att_bf);
    // 9) out = attn @ w_proj^T + b_proj (f32 -> d_out)  (8192x1024 K=1024)
    launch_gemm<1024, 2>(att_bf, wproj_bf, b_proj, outp, S, DIM, nullptr, stream);
}
